// SmallAxialUNet_49813030699177
// MI455X (gfx1250) — matrix operand order repaired and checked by reading, compiled, not run on hardware
//
#include <hip/hip_runtime.h>
#include <math.h>

// ---------------------------------------------------------------------------
// CDNA5 (gfx1250) types for WMMA
// ---------------------------------------------------------------------------
typedef __attribute__((ext_vector_type(16))) __bf16 v16bf;
typedef __attribute__((ext_vector_type(8)))  __bf16 v8bf;
typedef __attribute__((ext_vector_type(8)))  float  v8f;

// Native converts (gfx1250 has hardware f32->bf16; clang lowers the cast).
__device__ __forceinline__ __bf16 f2bf(float f) { return (__bf16)f; }
__device__ __forceinline__ unsigned short f2bfbits(float f) {
  __bf16 b = (__bf16)f;
  unsigned short h; __builtin_memcpy(&h, &b, 2); return h;
}
__device__ __forceinline__ __bf16 bfzero() {
  unsigned short h = 0; __bf16 b; __builtin_memcpy(&b, &h, 2); return b;
}

#define WMMA_BF16(va, vb, acc) \
  __builtin_amdgcn_wmma_f32_16x16x32_bf16(false, (va), false, (vb), (short)0, (acc), false, false)

// ---------------------------------------------------------------------------
// Weight pre-pad:  f32 [K,N] -> bf16 [Kp, Np] zero-padded (Kp%32==0, Np%64==0)
// ---------------------------------------------------------------------------
__global__ void k_pad_w(const float* __restrict__ w, __bf16* __restrict__ wp,
                        int K, int N, int Kp, int Np) {
  size_t total = (size_t)Kp * Np;
  size_t i = (size_t)blockIdx.x * blockDim.x + threadIdx.x;
  if (i >= total) return;
  int c = (int)(i % Np); int r = (int)(i / Np);
  float v = (r < K && c < N) ? w[(size_t)r * N + c] : 0.f;
  wp[i] = f2bf(v);
}

// ---------------------------------------------------------------------------
// A-operand loader for one 16x32 WMMA A tile.
// AFAST: K%32==0 -> unguarded float4 loads; else clamped-address selects.
// ---------------------------------------------------------------------------
template <bool AFAST>
__device__ __forceinline__ v16bf load_a16(const float* __restrict__ A, int arow,
                                          int K, int k0, int aoff) {
  v16bf va;
  if (AFAST) {
    const float* ap = A + (size_t)arow * K + k0 + aoff;
    float4 f0 = *(const float4*)(ap);
    float4 f1 = *(const float4*)(ap + 4);
    float4 f2 = *(const float4*)(ap + 16);
    float4 f3 = *(const float4*)(ap + 20);
    va[0]  = f2bf(f0.x); va[1]  = f2bf(f0.y); va[2]  = f2bf(f0.z); va[3]  = f2bf(f0.w);
    va[4]  = f2bf(f1.x); va[5]  = f2bf(f1.y); va[6]  = f2bf(f1.z); va[7]  = f2bf(f1.w);
    va[8]  = f2bf(f2.x); va[9]  = f2bf(f2.y); va[10] = f2bf(f2.z); va[11] = f2bf(f2.w);
    va[12] = f2bf(f3.x); va[13] = f2bf(f3.y); va[14] = f2bf(f3.z); va[15] = f2bf(f3.w);
  } else {
#pragma unroll
    for (int i = 0; i < 8; ++i) {
      int k1 = k0 + aoff + i, k2 = k0 + 16 + aoff + i;
      float f1 = A[(size_t)arow * K + ((k1 < K) ? k1 : 0)];
      float f2v = A[(size_t)arow * K + ((k2 < K) ? k2 : 0)];
      va[i]     = (k1 < K) ? f2bf(f1) : bfzero();
      va[i + 8] = (k2 < K) ? f2bf(f2v) : bfzero();
    }
  }
  return va;
}

// ---------------------------------------------------------------------------
// GEMM: C[M,N] = act(A[M,K] @ Wp[Kp,Np] (+bias)).  Wave computes 32x64 tile
// (M is always a multiple of 32 here), so each k-step amortizes the 4 B-tile
// loads over 8 WMMAs.
// ---------------------------------------------------------------------------
template <bool AFAST, bool BIAS, bool RELU>
__global__ __launch_bounds__(32)
void k_gemm(const float* __restrict__ A, const __bf16* __restrict__ Wp,
            const float* __restrict__ bias, float* __restrict__ C,
            int M, int K, int N, int Np) {
  (void)M;
  const int lane = threadIdx.x & 31;
  const int tn = blockIdx.x * 64;
  const int tm = blockIdx.y * 32;
  const int aoff = (lane < 16) ? 0 : 8;
  const int arow0 = tm + (lane & 15);
  const int arow1 = arow0 + 16;
  v8f acc[2][4];
  v8f zero = {};
#pragma unroll
  for (int h = 0; h < 2; ++h)
#pragma unroll
    for (int j = 0; j < 4; ++j) acc[h][j] = zero;

  for (int k0 = 0; k0 < K; k0 += 32) {
    v16bf va0 = load_a16<AFAST>(A, arow0, K, k0, aoff);
    v16bf va1 = load_a16<AFAST>(A, arow1, K, k0, aoff);
    if (AFAST && k0 + 32 < K)
      __builtin_prefetch(A + (size_t)arow0 * K + k0 + 32, 0, 3);
    const __bf16* wp = Wp + (size_t)k0 * Np + tn + (lane & 15);
#pragma unroll
    for (int j = 0; j < 4; ++j) {
      v16bf vb;
#pragma unroll
      for (int i = 0; i < 8; ++i) {
        vb[i]     = wp[(size_t)(aoff + i) * Np + j * 16];
        vb[i + 8] = wp[(size_t)(16 + aoff + i) * Np + j * 16];
      }
      acc[0][j] = WMMA_BF16(va0, vb, acc[0][j]);
      acc[1][j] = WMMA_BF16(va1, vb, acc[1][j]);
    }
  }

  const int n0 = tn + (lane & 15);
#pragma unroll
  for (int h = 0; h < 2; ++h) {
    const int mbase = tm + h * 16 + (lane < 16 ? 0 : 8);
#pragma unroll
    for (int j = 0; j < 4; ++j) {
      int n = n0 + j * 16;
      if (n < N) {
        float bb = BIAS ? bias[n] : 0.f;
#pragma unroll
        for (int r = 0; r < 8; ++r) {
          float v = acc[h][j][r] + bb;
          if (RELU) v = fmaxf(v, 0.f);
          C[(size_t)(mbase + r) * N + n] = v;
        }
      }
    }
  }
}

// ---------------------------------------------------------------------------
// Row attention per (sequence, head):  Q[S*L,hid], KV[S*L,2*hid] -> O[S*L,hid]
// Lp = NT*16 (L padded to 32).  K staged transposed [E][Lp], V [Lp][E] (bf16).
// Softmax with masking; unnormalized P@V streamed in 32-col chunks via a
// per-wave 16x32 LDS tile (intra-wave ordering: s_wait_dscnt).
// ---------------------------------------------------------------------------
template <int NT, int E>
__global__ __launch_bounds__(128)
void k_attn(const float* __restrict__ Q, const float* __restrict__ KV,
            float* __restrict__ O, int L, int heads, float scale) {
  constexpr int Lp = NT * 16;
  constexpr int NE = (E + 15) / 16;
  extern __shared__ char smem[];
  const int hid = heads * E;
  const int sh = blockIdx.x;
  const int s = sh / heads;
  const int h = sh - s * heads;
  const int tid = threadIdx.x;
  const int lane = tid & 31;
  const int wave = tid >> 5;
  const int nwaves = blockDim.x >> 5;

  __bf16* KshT = (__bf16*)smem;                     // [E][Lp]
  __bf16* Vsh  = KshT + (size_t)E * Lp;             // [Lp][E]
  __bf16* Pst  = Vsh + (size_t)Lp * E + wave * 512; // 16x32 per wave

  // ---- stage K^T and V ----
  if (E % 4 == 0) {
    const int ej4 = E / 4;
    for (int idx = tid; idx < Lp * ej4; idx += blockDim.x) {
      int n = idx / ej4;
      int j = (idx - n * ej4) * 4;
      int nc = (n < L) ? n : (L - 1);
      size_t bse = (size_t)(s * L + nc) * (size_t)(2 * hid) + (size_t)h * E + j;
      float4 kf = *(const float4*)(KV + bse);
      float4 vf = *(const float4*)(KV + bse + hid);
      if (n >= L) { kf = make_float4(0.f, 0.f, 0.f, 0.f); vf = kf; }
      KshT[(size_t)(j + 0) * Lp + n] = f2bf(kf.x);
      KshT[(size_t)(j + 1) * Lp + n] = f2bf(kf.y);
      KshT[(size_t)(j + 2) * Lp + n] = f2bf(kf.z);
      KshT[(size_t)(j + 3) * Lp + n] = f2bf(kf.w);
      union { unsigned short u[4]; uint2 v; } pk;
      pk.u[0] = f2bfbits(vf.x); pk.u[1] = f2bfbits(vf.y);
      pk.u[2] = f2bfbits(vf.z); pk.u[3] = f2bfbits(vf.w);
      *(uint2*)((unsigned short*)Vsh + (size_t)n * E + j) = pk.v;
    }
  } else {
    for (int idx = tid; idx < Lp * E; idx += blockDim.x) {
      int n = idx / E, j = idx - n * E;
      int nc = (n < L) ? n : (L - 1);
      size_t bse = (size_t)(s * L + nc) * (size_t)(2 * hid) + (size_t)h * E + j;
      float kf = KV[bse], vf = KV[bse + hid];
      if (n >= L) { kf = 0.f; vf = 0.f; }
      KshT[(size_t)j * Lp + n] = f2bf(kf);
      Vsh[(size_t)n * E + j] = f2bf(vf);
    }
  }
  __syncthreads();

  const int aoff = (lane < 16) ? 0 : 8;

  for (int qt = wave; qt < NT; qt += nwaves) {
    // ---- dots[16, Lp] = Q_tile @ K^T ----
    v8f dacc[NT];
    v8f zero = {};
#pragma unroll
    for (int t = 0; t < NT; ++t) dacc[t] = zero;

    const int qloc0 = qt * 16 + (lane & 15);
    const int qloc = (qloc0 < L) ? qloc0 : (L - 1);   // clamp (rows >= L unused)
    const size_t qbase = (size_t)(s * L + qloc) * hid + (size_t)h * E;

    for (int k0 = 0; k0 < E; k0 += 32) {
      v16bf va;
      if (E % 32 == 0) {
        const float* ap = Q + qbase + k0 + aoff;
        float4 f0 = *(const float4*)(ap);
        float4 f1 = *(const float4*)(ap + 4);
        float4 f2 = *(const float4*)(ap + 16);
        float4 f3 = *(const float4*)(ap + 20);
        va[0]  = f2bf(f0.x); va[1]  = f2bf(f0.y); va[2]  = f2bf(f0.z); va[3]  = f2bf(f0.w);
        va[4]  = f2bf(f1.x); va[5]  = f2bf(f1.y); va[6]  = f2bf(f1.z); va[7]  = f2bf(f1.w);
        va[8]  = f2bf(f2.x); va[9]  = f2bf(f2.y); va[10] = f2bf(f2.z); va[11] = f2bf(f2.w);
        va[12] = f2bf(f3.x); va[13] = f2bf(f3.y); va[14] = f2bf(f3.z); va[15] = f2bf(f3.w);
      } else {
#pragma unroll
        for (int i = 0; i < 8; ++i) {
          int k1 = k0 + aoff + i, k2 = k0 + 16 + aoff + i;
          float f1 = Q[qbase + ((k1 < E) ? k1 : 0)];
          float f2v = Q[qbase + ((k2 < E) ? k2 : 0)];
          va[i]     = (k1 < E) ? f2bf(f1) : bfzero();
          va[i + 8] = (k2 < E) ? f2bf(f2v) : bfzero();
        }
      }
#pragma unroll
      for (int t = 0; t < NT; ++t) {
        v16bf vb;
        if (E % 32 == 0) {
#pragma unroll
          for (int i = 0; i < 8; ++i) {
            vb[i]     = KshT[(size_t)(k0 + aoff + i) * Lp + t * 16 + (lane & 15)];
            vb[i + 8] = KshT[(size_t)(k0 + 16 + aoff + i) * Lp + t * 16 + (lane & 15)];
          }
        } else {
#pragma unroll
          for (int j = 0; j < 16; ++j) {
            const int kj = k0 + ((j < 8) ? (aoff + j) : (16 + aoff + j - 8));
            const int kc = (kj < E) ? kj : 0;
            __bf16 tv = KshT[(size_t)kc * Lp + t * 16 + (lane & 15)];
            vb[j] = (kj < E) ? tv : bfzero();
          }
        }
        dacc[t] = WMMA_BF16(va, vb, dacc[t]);
      }
    }

    // ---- scale, mask, row max (lanes<16 hold M=r, lanes>=16 hold M=8+r) ----
    float rowmax[8];
#pragma unroll
    for (int r = 0; r < 8; ++r) rowmax[r] = -3.0e38f;
#pragma unroll
    for (int t = 0; t < NT; ++t) {
      int n = t * 16 + (lane & 15);
#pragma unroll
      for (int r = 0; r < 8; ++r) {
        float v = dacc[t][r] * scale;
        if (n >= L) v = -3.0e38f;
        dacc[t][r] = v;
        rowmax[r] = fmaxf(rowmax[r], v);
      }
    }
#pragma unroll
    for (int r = 0; r < 8; ++r)
#pragma unroll
      for (int m = 1; m < 16; m <<= 1)
        rowmax[r] = fmaxf(rowmax[r], __shfl_xor(rowmax[r], m, 16));

    // ---- streaming exp + P@V over 32-column chunks ----
    float rowsum[8];
#pragma unroll
    for (int r = 0; r < 8; ++r) rowsum[r] = 0.f;
    v8f oacc[NE];
#pragma unroll
    for (int te = 0; te < NE; ++te) oacc[te] = zero;

#pragma unroll
    for (int c = 0; c < NT / 2; ++c) {
      const int mrow = (lane < 16) ? 0 : 8;
#pragma unroll
      for (int half = 0; half < 2; ++half) {
        const int t = 2 * c + half;
        const int col = half * 16 + (lane & 15);
#pragma unroll
        for (int r = 0; r < 8; ++r) {
          float p = __expf(dacc[t][r] - rowmax[r]);
          rowsum[r] += p;
          Pst[(mrow + r) * 32 + col] = f2bf(p);
        }
      }
      asm volatile("s_wait_dscnt 0" ::: "memory");  // LDS writes visible in-wave
      const int m = lane & 15;
      v8bf p0 = *(const v8bf*)(Pst + m * 32 + aoff);
      v8bf p1 = *(const v8bf*)(Pst + m * 32 + 16 + aoff);
      v16bf va = __builtin_shufflevector(p0, p1, 0, 1, 2, 3, 4, 5, 6, 7,
                                         8, 9, 10, 11, 12, 13, 14, 15);
      const int vrow = c * 32;
#pragma unroll
      for (int te = 0; te < NE; ++te) {
        v16bf vb;
        if (E % 16 == 0) {
#pragma unroll
          for (int j = 0; j < 16; ++j) {
            int kr = vrow + ((j < 8) ? (aoff + j) : (16 + aoff + j - 8));
            vb[j] = Vsh[(size_t)kr * E + te * 16 + (lane & 15)];
          }
        } else {
#pragma unroll
          for (int j = 0; j < 16; ++j) {
            int kr = vrow + ((j < 8) ? (aoff + j) : (16 + aoff + j - 8));
            int n = te * 16 + (lane & 15);
            __bf16 tv = Vsh[(size_t)kr * E + ((n < E) ? n : 0)];
            vb[j] = (n < E) ? tv : bfzero();
          }
        }
        oacc[te] = WMMA_BF16(va, vb, oacc[te]);
      }
    }

#pragma unroll
    for (int r = 0; r < 8; ++r)
#pragma unroll
      for (int m = 1; m < 16; m <<= 1)
        rowsum[r] += __shfl_xor(rowsum[r], m, 16);

    const int mloc = (lane < 16) ? 0 : 8;
#pragma unroll
    for (int te = 0; te < NE; ++te) {
      int n = te * 16 + (lane & 15);
      if (n >= E) continue;
#pragma unroll
      for (int r = 0; r < 8; ++r) {
        int qm = qt * 16 + mloc + r;
        if (qm < L)
          O[(size_t)(s * L + qm) * hid + (size_t)h * E + n] =
              oacc[te][r] / rowsum[r];
      }
    }
  }
}

// ---------------------------------------------------------------------------
// Elementwise / plumbing kernels
// ---------------------------------------------------------------------------
__global__ void k_zero(float* p, size_t n) {
  size_t i = (size_t)blockIdx.x * blockDim.x + threadIdx.x;
  if (i < n) p[i] = 0.f;
}

__global__ void k_transpose_hw(const float* __restrict__ in, float* __restrict__ out,
                               int H, int W, int C, size_t total) {
  size_t i = (size_t)blockIdx.x * blockDim.x + threadIdx.x;
  if (i >= total) return;
  int c = (int)(i % C); size_t t = i / C;
  int w = (int)(t % W); t /= W;
  int h = (int)(t % H); t /= H;
  int b = (int)t;
  out[(((size_t)b * W + w) * H + h) * C + c] = in[i];
}

__global__ void k_add_t_relu(const float* __restrict__ aw, const float* __restrict__ ahT,
                             float* __restrict__ out, int H, int W, int C,
                             size_t total) {
  size_t i = (size_t)blockIdx.x * blockDim.x + threadIdx.x;
  if (i >= total) return;
  int c = (int)(i % C); size_t t = i / C;
  int w = (int)(t % W); t /= W;
  int h = (int)(t % H); t /= H;
  int b = (int)t;
  float v = aw[i] + ahT[(((size_t)b * W + w) * H + h) * C + c];
  out[i] = fmaxf(v, 0.f);
}

__global__ void k_concat3(const float* __restrict__ a, int ca,
                          const float* __restrict__ b, int cb,
                          const float* __restrict__ c, int cc,
                          float* __restrict__ out, size_t ntok) {
  int ct = ca + cb + cc;
  size_t total = ntok * (size_t)ct;
  size_t i = (size_t)blockIdx.x * blockDim.x + threadIdx.x;
  if (i >= total) return;
  int ch = (int)(i % ct); size_t n = i / ct;
  float v;
  if (ch < ca) v = a[n * ca + ch];
  else if (ch < ca + cb) v = b[n * cb + (ch - ca)];
  else v = c[n * cc + (ch - ca - cb)];
  out[i] = v;
}

__global__ void k_maxpool2(const float* __restrict__ in, float* __restrict__ out,
                           int H, int W, int C, size_t total) {
  int Ho = H / 2, Wo = W / 2;
  size_t i = (size_t)blockIdx.x * blockDim.x + threadIdx.x;
  if (i >= total) return;
  int c = (int)(i % C); size_t t = i / C;
  int w = (int)(t % Wo); t /= Wo;
  int h = (int)(t % Ho); t /= Ho;
  int b = (int)t;
  const float* p = in + (((size_t)b * H + 2 * h) * W + 2 * w) * C + c;
  size_t rs = (size_t)W * C;
  out[i] = fmaxf(fmaxf(p[0], p[C]), fmaxf(p[rs], p[rs + C]));
}

__global__ void k_upsample2(const float* __restrict__ in, float* __restrict__ out,
                            int H, int W, int C, size_t total) {
  int Ho = 2 * H, Wo = 2 * W;
  size_t i = (size_t)blockIdx.x * blockDim.x + threadIdx.x;
  if (i >= total) return;
  int c = (int)(i % C); size_t t = i / C;
  int wo = (int)(t % Wo); t /= Wo;
  int ho = (int)(t % Ho); t /= Ho;
  int b = (int)t;
  float ry = (float)(H - 1) / (float)(Ho - 1);
  float rx = (float)(W - 1) / (float)(Wo - 1);
  float py = ho * ry, px = wo * rx;
  int y0 = (int)floorf(py), x0 = (int)floorf(px);
  int y1 = min(y0 + 1, H - 1), x1 = min(x0 + 1, W - 1);
  float wy = py - (float)y0, wx = px - (float)x0;
  auto at = [&](int y, int x) {
    return in[(((size_t)b * H + y) * W + x) * C + c];
  };
  out[i] = (1.f - wy) * ((1.f - wx) * at(y0, x0) + wx * at(y0, x1)) +
           wy * ((1.f - wx) * at(y1, x0) + wx * at(y1, x1));
}

// blockDim.x == C; each block reduces a token chunk, two atomics (sum, sum of squares) per channel.
__global__ void k_bn_stats(const float* __restrict__ x, float* __restrict__ st,
                           size_t ntok, int C, size_t chunk) {
  int c = threadIdx.x;
  size_t t0 = (size_t)blockIdx.x * chunk;
  size_t t1 = t0 + chunk; if (t1 > ntok) t1 = ntok;
  float s = 0.f, q = 0.f;
  for (size_t t = t0; t < t1; ++t) {
    float v = x[t * C + c];
    s += v; q += v * v;
  }
  atomicAdd(&st[c], s);
  atomicAdd(&st[C + c], q);
}

__global__ void k_bn_apply(const float* __restrict__ x, const float* __restrict__ st,
                           const float* __restrict__ g, const float* __restrict__ b,
                           float* __restrict__ y, size_t ntok, int C) {
  size_t total = ntok * (size_t)C;
  size_t i = (size_t)blockIdx.x * blockDim.x + threadIdx.x;
  if (i >= total) return;
  int c = (int)(i % C);
  float inv_n = 1.0f / (float)ntok;
  float mu = st[c] * inv_n;
  float var = st[C + c] * inv_n - mu * mu;
  y[i] = g[c] * (x[i] - mu) * rsqrtf(var + 1e-5f) + b[c];
}

// ---------------------------------------------------------------------------
// Host orchestration
// ---------------------------------------------------------------------------
static inline unsigned g1d(size_t n) { return (unsigned)((n + 255) / 256); }

struct PW { const __bf16* p; int Kp, Np; };

static PW prep_w(hipStream_t st, const float* w, int K, int N,
                 char* base, size_t& off) {
  int Kp = (K + 31) & ~31;
  int Np = (N + 63) & ~63;
  __bf16* d = (__bf16*)(base + off);
  size_t bytes = (size_t)Kp * Np * sizeof(__bf16);
  off = (off + bytes + 255) & ~(size_t)255;
  k_pad_w<<<g1d((size_t)Kp * Np), 256, 0, st>>>(w, d, K, N, Kp, Np);
  return {d, Kp, Np};
}

static void gemm(hipStream_t st, const float* A, const PW& w, const float* bias,
                 float* C, size_t M, int K, int N, bool relu) {
  dim3 grid((unsigned)(w.Np / 64), (unsigned)((M + 31) / 32));
  if (K % 32 == 0) {
    if (bias)      k_gemm<true, true, false><<<grid, 32, 0, st>>>(A, w.p, bias, C, (int)M, K, N, w.Np);
    else if (relu) k_gemm<true, false, true><<<grid, 32, 0, st>>>(A, w.p, nullptr, C, (int)M, K, N, w.Np);
    else           k_gemm<true, false, false><<<grid, 32, 0, st>>>(A, w.p, nullptr, C, (int)M, K, N, w.Np);
  } else {
    if (bias)      k_gemm<false, true, false><<<grid, 32, 0, st>>>(A, w.p, bias, C, (int)M, K, N, w.Np);
    else if (relu) k_gemm<false, false, true><<<grid, 32, 0, st>>>(A, w.p, nullptr, C, (int)M, K, N, w.Np);
    else           k_gemm<false, false, false><<<grid, 32, 0, st>>>(A, w.p, nullptr, C, (int)M, K, N, w.Np);
  }
}

static void launch_attn(hipStream_t st, const float* q, const float* kv, float* ctx,
                        int S, int L, int heads, int e) {
  const int Lp = ((L + 31) / 32) * 32;
  const int NT = Lp / 16;
  const float scale = 1.0f / sqrtf((float)e);
  const size_t shbytes = (size_t)(2 * (size_t)Lp * e + 4 * 512) * 2;
  dim3 grid((unsigned)(S * heads));
#define ATTN_LAUNCH(nt, ee)                                                        \
  else if (NT == nt && e == ee) {                                                  \
    hipFuncSetAttribute(reinterpret_cast<const void*>(&k_attn<nt, ee>),            \
                        hipFuncAttributeMaxDynamicSharedMemorySize, (int)shbytes); \
    k_attn<nt, ee><<<grid, 128, shbytes, st>>>(q, kv, ctx, L, heads, scale);       \
  }
  if (false) {}
  ATTN_LAUNCH(14, 1)
  ATTN_LAUNCH(8, 32)
  ATTN_LAUNCH(4, 64)
  ATTN_LAUNCH(8, 128)
  ATTN_LAUNCH(14, 64)
#undef ATTN_LAUNCH
}

struct BlockP {
  PW qw, kvw, ow, qh, kvh, oh, conv;
  const float *bow, *boh, *bng, *bnb;
};

static void self_attn(hipStream_t st, const float* x, float* out, int S, int L,
                      int C, int heads, int e, const PW& wq, const PW& wkv,
                      const PW& wo, const float* bo, float* q, float* kv,
                      float* ctx) {
  const int hid = heads * e;
  const size_t M = (size_t)S * L;
  gemm(st, x, wq, nullptr, q, M, C, hid, false);
  gemm(st, x, wkv, nullptr, kv, M, C, 2 * hid, false);
  launch_attn(st, q, kv, ctx, S, L, heads, e);
  gemm(st, ctx, wo, bo, out, M, hid, C, false);
}

static void run_block(hipStream_t st, const float* xin, const float* extra,
                      float* out, int Bb, int Hh, int Ww, int Cin, int Cext,
                      int Cout, int e, const BlockP& w, float* q, float* kv,
                      float* ctx, float* aw, float* ah, float* xt, float* cat,
                      float* stats) {
  const int heads = 2;
  const size_t M = (size_t)Bb * Hh * Ww;
  // attention along W
  self_attn(st, xin, aw, Bb * Hh, Ww, Cin, heads, e, w.qw, w.kvw, w.ow, w.bow,
            q, kv, ctx);
  // attention along H (on transposed input)
  size_t totC = M * (size_t)Cin;
  k_transpose_hw<<<g1d(totC), 256, 0, st>>>(xin, xt, Hh, Ww, Cin, totC);
  self_attn(st, xt, ah, Bb * Ww, Hh, Cin, heads, e, w.qh, w.kvh, w.oh, w.boh,
            q, kv, ctx);
  // a = relu(aw + ah^T)
  k_add_t_relu<<<g1d(totC), 256, 0, st>>>(aw, ah, aw, Hh, Ww, Cin, totC);
  // concat [a, x, extra]
  int ct = 2 * Cin + Cext;
  k_concat3<<<g1d(M * (size_t)ct), 256, 0, st>>>(aw, Cin, xin, Cin, extra, Cext,
                                                 cat, M);
  // 1x1 conv + relu
  gemm(st, cat, w.conv, nullptr, out, M, ct, Cout, true);
  // BatchNorm (training-mode stats)
  k_zero<<<g1d(512), 256, 0, st>>>(stats, 512);
  size_t chunk = (M + 255) / 256;
  k_bn_stats<<<256, Cout, 0, st>>>(out, stats, M, Cout, chunk);
  k_bn_apply<<<g1d(M * (size_t)Cout), 256, 0, st>>>(out, stats, w.bng, w.bnb,
                                                    out, M, Cout);
}

extern "C" void kernel_launch(void* const* d_in, const int* in_sizes, int n_in,
                              void* d_out, int out_size, void* d_ws, size_t ws_size,
                              hipStream_t stream) {
  (void)out_size; (void)ws_size;
  if (n_in < 57) return;
  const float* const* in = (const float* const*)d_in;

  struct RawB { const float *wq_w,*wkv_w,*wo_w,*bo_w,*wq_h,*wkv_h,*wo_h,*bo_h,*conv,*bng,*bnb; };
  const float* x;
  const float* decode_w;
  RawB rb[5];
  // Detect pytree flattening order: insertion order puts x (602112 elems) first.
  const bool ins = (in_sizes[0] == 4 * 224 * 224 * 3);
  if (ins) {
    x = in[0];
    for (int i = 0; i < 5; ++i) {
      int b = 1 + i * 11;
      rb[i] = {in[b+0], in[b+1], in[b+2], in[b+3], in[b+4], in[b+5], in[b+6],
               in[b+7], in[b+8], in[b+9], in[b+10]};
    }
    decode_w = in[56];
  } else {
    // jax sorted-key order: decode_w, then blocks (down1,down2,encode,up1,up2)
    // each {attn{ax_h{bo,wkv,wo,wq}, ax_w{bo,wkv,wo,wq}}, bn_b, bn_g, conv_w}, then x.
    decode_w = in[0];
    const int pos[5] = {2, 0, 1, 3, 4};  // our order: encode,down1,down2,up1,up2
    for (int i = 0; i < 5; ++i) {
      int b = 1 + pos[i] * 11;
      rb[i] = {in[b+7], in[b+5], in[b+6], in[b+4],   // ax_w: wq,wkv,wo,bo
               in[b+3], in[b+1], in[b+2], in[b+0],   // ax_h: wq,wkv,wo,bo
               in[b+10], in[b+9], in[b+8]};          // conv_w, bn_g, bn_b
    }
    x = in[56];
  }

  // ---- workspace layout (bytes) ----
  char* base = (char*)d_ws;
  size_t off = 0;
  auto allocf = [&](size_t n) {
    float* p = (float*)(base + off);
    off = (off + n * sizeof(float) + 255) & ~(size_t)255;
    return p;
  };
  const int Bb = 4;
  const size_t N1 = (size_t)Bb * 224 * 224;
  const size_t N2 = (size_t)Bb * 112 * 112;
  const size_t N3 = (size_t)Bb * 56 * 56;

  float* x1  = allocf(N1 * 64);
  float* x1p = allocf(N2 * 64);
  float* x2  = allocf(N2 * 128);
  float* x2p = allocf(N3 * 128);
  float* x3  = allocf(N3 * 256);
  float* x3u = allocf(N2 * 256);
  float* x4  = allocf(N2 * 128);
  float* x4u = allocf(N1 * 128);
  float* x5  = allocf(N1 * 64);
  float* q   = allocf(N1 * 128);
  float* kv  = allocf(N1 * 256);
  float* ctx = allocf(N1 * 128);
  float* aw  = allocf(N1 * 128);
  float* ah  = allocf(N1 * 128);
  float* xt  = allocf(N1 * 128);
  float* cat = allocf(N1 * 320);
  float* stats = allocf(512);

  // ---- pre-pad weights to bf16 ----
  const int cin[5]  = {3, 64, 128, 256, 128};
  const int ee[5]   = {1, 32, 64, 128, 64};
  const int cext[5] = {0, 0, 0, 128, 64};
  const int cout[5] = {64, 128, 256, 128, 64};
  BlockP bw[5];
  for (int i = 0; i < 5; ++i) {
    int hid = 2 * ee[i];
    bw[i].qw   = prep_w(stream, rb[i].wq_w,  cin[i], hid,     base, off);
    bw[i].kvw  = prep_w(stream, rb[i].wkv_w, cin[i], 2 * hid, base, off);
    bw[i].ow   = prep_w(stream, rb[i].wo_w,  hid,    cin[i],  base, off);
    bw[i].qh   = prep_w(stream, rb[i].wq_h,  cin[i], hid,     base, off);
    bw[i].kvh  = prep_w(stream, rb[i].wkv_h, cin[i], 2 * hid, base, off);
    bw[i].oh   = prep_w(stream, rb[i].wo_h,  hid,    cin[i],  base, off);
    bw[i].conv = prep_w(stream, rb[i].conv, 2 * cin[i] + cext[i], cout[i], base, off);
    bw[i].bow = rb[i].bo_w; bw[i].boh = rb[i].bo_h;
    bw[i].bng = rb[i].bng;  bw[i].bnb = rb[i].bnb;
  }
  PW dec = prep_w(stream, decode_w, 64, 2, base, off);

  // ---- forward pass ----
  run_block(stream, x,   nullptr, x1, Bb, 224, 224,   3,   0,  64,   1, bw[0],
            q, kv, ctx, aw, ah, xt, cat, stats);
  k_maxpool2<<<g1d(N2 * 64), 256, 0, stream>>>(x1, x1p, 224, 224, 64, N2 * 64);
  run_block(stream, x1p, nullptr, x2, Bb, 112, 112,  64,   0, 128,  32, bw[1],
            q, kv, ctx, aw, ah, xt, cat, stats);
  k_maxpool2<<<g1d(N3 * 128), 256, 0, stream>>>(x2, x2p, 112, 112, 128, N3 * 128);
  run_block(stream, x2p, nullptr, x3, Bb,  56,  56, 128,   0, 256,  64, bw[2],
            q, kv, ctx, aw, ah, xt, cat, stats);
  k_upsample2<<<g1d(N2 * 256), 256, 0, stream>>>(x3, x3u, 56, 56, 256, N2 * 256);
  run_block(stream, x3u, x2,     x4, Bb, 112, 112, 256, 128, 128, 128, bw[3],
            q, kv, ctx, aw, ah, xt, cat, stats);
  k_upsample2<<<g1d(N1 * 128), 256, 0, stream>>>(x4, x4u, 112, 112, 128, N1 * 128);
  run_block(stream, x4u, x1,     x5, Bb, 224, 224, 128,  64,  64,  64, bw[4],
            q, kv, ctx, aw, ah, xt, cat, stats);
  // decode: (N1,64) @ (64,2)
  gemm(stream, x5, dec, nullptr, (float*)d_out, N1, 64, 2, false);
}
